// MogLSTM_12790412607595
// MI455X (gfx1250) — compile-verified, matrix operand order checked by reading, not run on hardware
//
#include <hip/hip_runtime.h>

// ---------------------------------------------------------------------------
// MogLSTM (bidirectional mogrifier LSTM) for gfx1250.
// B=128, T=512, D=H=256.  bf16 WMMA (v_wmma_f32_16x16x32_bf16) for all GEMMs.
// 16 persistent workgroups (2 dirs x 8 batch tiles of 16 rows), state in LDS,
// weights streamed from L2 (hoisted to VGPRs by the compiler), x tiles double-
// buffered into LDS with GLOBAL_LOAD_ASYNC_TO_LDS_B128 (ASYNCcnt), outputs
// accumulated with global_atomic_add_f32 (2 deterministic contributions/elem).
// ---------------------------------------------------------------------------

namespace {
constexpr int Bn = 128;
constexpr int Tn = 512;
constexpr int Dn = 256;
constexpr int Hn = 256;
constexpr int H4 = 1024;
constexpr int SX = Dn + 8;   // LDS row stride for bf16 tiles (528B, 16B aligned)
constexpr int SC = Hn + 4;   // LDS row stride for f32 c-state
}

typedef __bf16 bf16_t;
typedef __bf16 v16bf __attribute__((ext_vector_type(16)));
typedef __bf16 v8bf  __attribute__((ext_vector_type(8)));
typedef float  v8f   __attribute__((ext_vector_type(8)));

__device__ __forceinline__ float sigf(float x) { return 1.0f / (1.0f + __expf(-x)); }

// Load a 16x32 bf16 WMMA fragment half for this lane.
// Layout (ISA 7.12.2, 16-bit A 16x32; B mirrored with N in place of M):
//   lane row = lane&15, k-offset = 8*(lane>=16)
//   vector elems 0..7  = K[k0+koff .. +7]       (16 contiguous bytes)
//   vector elems 8..15 = K[k0+16+koff .. +7]    (16 contiguous bytes)
__device__ __forceinline__ v16bf ldfrag(const bf16_t* p) {
  v8bf lo = *(const v8bf*)(p);
  v8bf hi = *(const v8bf*)(p + 16);
  return __builtin_shufflevector(lo, hi, 0, 1, 2, 3, 4, 5, 6, 7,
                                 8, 9, 10, 11, 12, 13, 14, 15);
}

__device__ __forceinline__ v8f wmma_bf16(v16bf a, v16bf b, v8f c) {
  return __builtin_amdgcn_wmma_f32_16x16x32_bf16(false, a, false, b,
                                                 (short)0, c, false, false);
}

// Async global->LDS copy of one 16-byte chunk (8 bf16). Tracked by ASYNCcnt.
// VDST = LDS byte offset (generic shared pointer low 32 bits; the LDS aperture
// occupies addr[63:32] only), VADDR = 64-bit global address, GV mode.
__device__ __forceinline__ void async_ld16(bf16_t* lds, const bf16_t* g) {
  const unsigned ldsoff = (unsigned)(size_t)lds;
  asm volatile("global_load_async_to_lds_b128 %0, %1, off"
               :: "v"(ldsoff), "v"(g));
}

// Prefetch one 16x256 bf16 x tile (8 KB) into LDS: 512 async b128 ops,
// 2 per thread (256 threads).
__device__ __forceinline__ void prefetch_x(bf16_t* buf, const bf16_t* xbf,
                                           int b0, int ta, int tid) {
#pragma unroll
  for (int i = tid; i < 512; i += 256) {
    const int m = i >> 5;          // 32 chunks of 8 elems per row
    const int d = (i & 31) << 3;
    async_ld16(buf + m * SX + d,
               xbf + ((size_t)(b0 + m) * Tn + ta) * Dn + d);
  }
}

// One mogrifier stage: S = shA(16xK) @ Bg(KxN), shD *= 2*sigmoid(S), K=N=256.
// Each wave owns 2 N-tiles of 16 columns; 8 K-steps of 32 per tile.
__device__ __forceinline__ void mog_stage(const bf16_t* shA, const bf16_t* Bg,
                                          bf16_t* shD, int wave, int lane) {
  const int nloc = lane & 15;
  const int moff = (lane >> 4) << 3;  // C/D row offset AND A/B k offset
  const int mA   = lane & 15;
  for (int nt = 2 * wave; nt < 2 * wave + 2; ++nt) {
    const bf16_t* brow = Bg + (size_t)(nt * 16 + nloc) * 256;
    v8f acc = {};
#pragma unroll
    for (int k0 = 0; k0 < 256; k0 += 32) {
      v16bf a = ldfrag(&shA[mA * SX + k0 + moff]);
      v16bf b = ldfrag(brow + k0 + moff);
      acc = wmma_bf16(a, b, acc);
    }
#pragma unroll
    for (int r = 0; r < 8; ++r) {
      const int m = r + moff;
      const int n = nt * 16 + nloc;
      const float sc = 2.0f * sigf(acc[r]);
      bf16_t* pd = &shD[m * SX + n];
      *pd = (bf16_t)(sc * (float)(*pd));
    }
  }
}

// ---------------------------------------------------------------------------
// Prep: fp32 -> bf16 conversion of x and weights (Q/R transposed so that the
// B-operand K-walk is memory-contiguous per output column), bias fusion,
// output zeroing.
// ---------------------------------------------------------------------------
__global__ void moglstm_prep(const float* __restrict__ x,
                             const float* __restrict__ Q,
                             const float* __restrict__ R,
                             const float* __restrict__ Wih_fw,
                             const float* __restrict__ Whh_fw,
                             const float* __restrict__ bih_fw,
                             const float* __restrict__ bhh_fw,
                             const float* __restrict__ Wih_bw,
                             const float* __restrict__ Whh_bw,
                             const float* __restrict__ bih_bw,
                             const float* __restrict__ bhh_bw,
                             bf16_t* __restrict__ xbf,
                             bf16_t* __restrict__ Wihbf,
                             bf16_t* __restrict__ Whhbf,
                             bf16_t* __restrict__ QT,
                             bf16_t* __restrict__ RT,
                             float* __restrict__ bias,
                             float* __restrict__ out) {
  const size_t stride = (size_t)gridDim.x * blockDim.x;
  const size_t t0 = (size_t)blockIdx.x * blockDim.x + threadIdx.x;

  for (size_t i = t0; i < (size_t)Bn * Tn * Dn; i += stride) xbf[i] = (bf16_t)x[i];

  for (size_t i = t0; i < (size_t)H4 * Dn; i += stride) {
    Wihbf[i] = (bf16_t)Wih_fw[i];
    Wihbf[(size_t)H4 * Dn + i] = (bf16_t)Wih_bw[i];
  }
  for (size_t i = t0; i < (size_t)H4 * Hn; i += stride) {
    Whhbf[i] = (bf16_t)Whh_fw[i];
    Whhbf[(size_t)H4 * Hn + i] = (bf16_t)Whh_bw[i];
  }
  // QT[dir][n*H + k] = Q[dir][k*D + n]   (h @ Q : K=H, N=D)
  for (size_t i = t0; i < (size_t)Dn * Hn; i += stride) {
    const size_t n = i / Hn, k = i % Hn;
    QT[i] = (bf16_t)Q[k * Dn + n];
    QT[(size_t)Dn * Hn + i] = (bf16_t)Q[(size_t)Hn * Dn + k * Dn + n];
  }
  // RT[dir][n*D + k] = R[dir][k*H + n]   (x @ R : K=D, N=H)
  for (size_t i = t0; i < (size_t)Hn * Dn; i += stride) {
    const size_t n = i / Dn, k = i % Dn;
    RT[i] = (bf16_t)R[k * Hn + n];
    RT[(size_t)Hn * Dn + i] = (bf16_t)R[(size_t)Dn * Hn + k * Hn + n];
  }
  for (size_t i = t0; i < (size_t)H4; i += stride) {
    bias[i] = bih_fw[i] + bhh_fw[i];
    bias[H4 + i] = bih_bw[i] + bhh_bw[i];
  }
  for (size_t i = t0; i < (size_t)Bn * Tn * Hn; i += stride) out[i] = 0.0f;
}

// ---------------------------------------------------------------------------
// Persistent scan kernel: blockIdx.x = dir*8 + batch_tile.
// ---------------------------------------------------------------------------
__global__ __launch_bounds__(256)
void moglstm_scan(const bf16_t* __restrict__ xbf,
                  const int* __restrict__ lengths,
                  const bf16_t* __restrict__ WihAll,   // [2][4H][D]
                  const bf16_t* __restrict__ WhhAll,   // [2][4H][H]
                  const bf16_t* __restrict__ QTall,    // [2][D][H]
                  const bf16_t* __restrict__ RTall,    // [2][H][D]
                  const float* __restrict__ biasAll,   // [2][4H]
                  float* __restrict__ out) {           // [B][T][H]
  __shared__ __align__(16) bf16_t sh_h [16 * SX];      // hidden state, bf16
  __shared__ __align__(16) bf16_t sh_hm[16 * SX];      // mogrified h
  __shared__ __align__(16) bf16_t sh_xp[2][16 * SX];   // double-buffered x / xm
  __shared__ __align__(16) float  sh_c [16 * SC];      // cell state, f32

  const int dir  = blockIdx.x >> 3;
  const int b0   = (blockIdx.x & 7) * 16;
  const int tid  = threadIdx.x;
  const int wave = tid >> 5;
  const int lane = tid & 31;
  const int nloc = lane & 15;
  const int moff = (lane >> 4) << 3;
  const int mA   = lane & 15;

  const bf16_t* Wih = WihAll + (size_t)dir * H4 * Dn;
  const bf16_t* Whh = WhhAll + (size_t)dir * H4 * Hn;
  const bf16_t* Qt  = QTall + (size_t)dir * Dn * Hn;
  const bf16_t* Rt  = RTall + (size_t)dir * Hn * Dn;
  const float*  bsp = biasAll + (size_t)dir * H4;

  int lenr[8];
#pragma unroll
  for (int r = 0; r < 8; ++r) lenr[r] = lengths[b0 + moff + r];

  for (int i = tid; i < 16 * SX; i += 256) sh_h[i] = (bf16_t)0.0f;
  for (int i = tid; i < 16 * SC; i += 256) sh_c[i] = 0.0f;

  // Prologue: async-prefetch the t=0 x tile into buffer 0.
  prefetch_x(sh_xp[0], xbf, b0, dir ? (Tn - 1) : 0, tid);
  __syncthreads();

  for (int t = 0; t < Tn; ++t) {
    const int ta = dir ? (Tn - 1 - t) : t;
    bf16_t* sh_xm = sh_xp[t & 1];

    // Stage 1: kick off async prefetch of the NEXT x tile into the other
    // buffer (its readers finished at the previous end-of-step barrier),
    // then wait only for the CURRENT tile (async loads retire in order, so
    // asynccnt<=2 leaves exactly this thread's 2 new ops in flight).
    if (t + 1 < Tn) {
      prefetch_x(sh_xp[(t + 1) & 1], xbf, b0, dir ? (Tn - 2 - t) : (t + 1), tid);
      asm volatile("s_wait_asynccnt 2");
    } else {
      asm volatile("s_wait_asynccnt 0");
    }
    // hm = h (LDS -> LDS)
    for (int i = tid; i < 16 * Dn; i += 256) {
      const int m = i >> 8;
      const int d = i & (Dn - 1);
      sh_hm[m * SX + d] = sh_h[m * SX + d];
    }
    __syncthreads();

    // Mogrifier: 2 rounds of (x-update, h-update)
    mog_stage(sh_hm, Qt, sh_xm, wave, lane); __syncthreads();
    mog_stage(sh_xm, Rt, sh_hm, wave, lane); __syncthreads();
    mog_stage(sh_hm, Qt, sh_xm, wave, lane); __syncthreads();
    mog_stage(sh_xm, Rt, sh_hm, wave, lane); __syncthreads();

    // Gates: xm @ Wih^T + hm @ Whh^T, then LSTM cell update.
    // Each wave owns 2 slices of 16 h-columns; 4 independent gate accum chains.
    for (int js = 2 * wave; js < 2 * wave + 2; ++js) {
      const int ng = js * 16 + nloc;  // this lane's h column
      const bf16_t* wi0 = Wih + (size_t)(0 * Hn + ng) * Dn;
      const bf16_t* wi1 = Wih + (size_t)(1 * Hn + ng) * Dn;
      const bf16_t* wi2 = Wih + (size_t)(2 * Hn + ng) * Dn;
      const bf16_t* wi3 = Wih + (size_t)(3 * Hn + ng) * Dn;
      const bf16_t* wh0 = Whh + (size_t)(0 * Hn + ng) * Hn;
      const bf16_t* wh1 = Whh + (size_t)(1 * Hn + ng) * Hn;
      const bf16_t* wh2 = Whh + (size_t)(2 * Hn + ng) * Hn;
      const bf16_t* wh3 = Whh + (size_t)(3 * Hn + ng) * Hn;

      v8f a0 = {}, a1 = {}, a2 = {}, a3 = {};
#pragma unroll
      for (int k0 = 0; k0 < Dn; k0 += 32) {
        v16bf ax = ldfrag(&sh_xm[mA * SX + k0 + moff]);
        v16bf ah = ldfrag(&sh_hm[mA * SX + k0 + moff]);
        a0 = wmma_bf16(ax, ldfrag(wi0 + k0 + moff), a0);
        a1 = wmma_bf16(ax, ldfrag(wi1 + k0 + moff), a1);
        a2 = wmma_bf16(ax, ldfrag(wi2 + k0 + moff), a2);
        a3 = wmma_bf16(ax, ldfrag(wi3 + k0 + moff), a3);
        a0 = wmma_bf16(ah, ldfrag(wh0 + k0 + moff), a0);
        a1 = wmma_bf16(ah, ldfrag(wh1 + k0 + moff), a1);
        a2 = wmma_bf16(ah, ldfrag(wh2 + k0 + moff), a2);
        a3 = wmma_bf16(ah, ldfrag(wh3 + k0 + moff), a3);
      }

      const float bi = bsp[0 * Hn + ng];
      const float bf_ = bsp[1 * Hn + ng];
      const float bg = bsp[2 * Hn + ng];
      const float bo = bsp[3 * Hn + ng];
#pragma unroll
      for (int r = 0; r < 8; ++r) {
        const int m = r + moff;
        const float gi = sigf(a0[r] + bi);
        const float gf = sigf(a1[r] + bf_);
        const float gg = tanhf(a2[r] + bg);
        const float go = sigf(a3[r] + bo);
        const float cold = sh_c[m * SC + ng];
        const float cn = gf * cold + gi * gg;
        const float hn = go * tanhf(cn);
        if (ta < lenr[r]) {  // sequence mask: keep old state when inactive
          sh_c[m * SC + ng] = cn;
          sh_h[m * SX + ng] = (bf16_t)hn;
          // fw + bw each add exactly once per element -> deterministic sum
          atomicAdd(&out[((size_t)(b0 + m) * Tn + ta) * Hn + ng], hn);
        }
      }
    }
    __syncthreads();
  }
}

// ---------------------------------------------------------------------------
extern "C" void kernel_launch(void* const* d_in, const int* in_sizes, int n_in,
                              void* d_out, int out_size, void* d_ws, size_t ws_size,
                              hipStream_t stream) {
  const float* x      = (const float*)d_in[0];
  const int*   lens   = (const int*)d_in[1];
  const float* Q      = (const float*)d_in[2];
  const float* R      = (const float*)d_in[3];
  const float* Wih_fw = (const float*)d_in[4];
  const float* Whh_fw = (const float*)d_in[5];
  const float* bih_fw = (const float*)d_in[6];
  const float* bhh_fw = (const float*)d_in[7];
  const float* Wih_bw = (const float*)d_in[8];
  const float* Whh_bw = (const float*)d_in[9];
  const float* bih_bw = (const float*)d_in[10];
  const float* bhh_bw = (const float*)d_in[11];
  float* outf = (float*)d_out;

  // Workspace layout (bf16 staging, ~36.2 MB total)
  bf16_t* xbf   = (bf16_t*)d_ws;                       // B*T*D
  bf16_t* Wihbf = xbf + (size_t)Bn * Tn * Dn;          // 2*4H*D
  bf16_t* Whhbf = Wihbf + (size_t)2 * H4 * Dn;         // 2*4H*H
  bf16_t* QTp   = Whhbf + (size_t)2 * H4 * Hn;         // 2*D*H
  bf16_t* RTp   = QTp + (size_t)2 * Dn * Hn;           // 2*H*D
  float*  biasp = (float*)(RTp + (size_t)2 * Hn * Dn); // 2*4H f32

  moglstm_prep<<<2048, 256, 0, stream>>>(x, Q, R, Wih_fw, Whh_fw, bih_fw, bhh_fw,
                                         Wih_bw, Whh_bw, bih_bw, bhh_bw,
                                         xbf, Wihbf, Whhbf, QTp, RTp, biasp, outf);

  moglstm_scan<<<16, 256, 0, stream>>>(xbf, lens, Wihbf, Whhbf, QTp, RTp, biasp, outf);
}
